// Preprocess_Coarse2Fine_64072322122004
// MI455X (gfx1250) — matrix operand order repaired and checked by reading, not compiled, not run on hardware
//
#include <hip/hip_runtime.h>

// CAPS "expected correspondence within window" for MI455X (gfx1250, wave32).
// Sizes fixed by setup_inputs(): b=2, n_pts=2048, d=128, H=120, W=160,
// window 15x20 = 300 samples. Output = [expected(2,2048,2) | std(2,2048)] fp32.

typedef __attribute__((ext_vector_type(16))) _Float16 v16h;
typedef __attribute__((ext_vector_type(8)))  _Float16 h8;
typedef __attribute__((ext_vector_type(8)))  float    v8f;

#define NB     2
#define NPTS   2048
#define DCH    128
#define H2     120
#define W2     160
#define LEN_H  15
#define LEN_W  20
#define KWIN   300          // LEN_H * LEN_W
#define PATCH_W 22          // 20px x-span -> 22 columns (incl. +1 corner)
#define PATCH_H 17          // 15px y-span -> 17 rows
#define NPIX   374          // 22*17
#define TILES  24           // ceil(374/16)
#define GSTRIDE 384         // TILES*16

#define NPIXELS   (NB * H2 * W2)   // 38400 real pixels
#define ZPIX      NPIXELS          // guard pixel index (always zero channels)
// workspace layout (halfs): FT [0 .. (NPIXELS+1)*128) ; feat1-f16 after that
#define F1H_OFF   ((size_t)(NPIXELS + 1) * DCH)

// ---------------------------------------------------------------------------
// Pass 1: transpose + convert featmap2 [b,128,H,W] f32 -> FT [b,H,W,128] f16
// ---------------------------------------------------------------------------
__global__ __launch_bounds__(256) void ft_transpose_kernel(
    const float* __restrict__ fm, _Float16* __restrict__ ft) {
  int tid = blockIdx.x * 256 + threadIdx.x;       // ((bi*16+cg)*H + y)*W + x
  if (tid >= NB * 16 * H2 * W2) return;
  int x  = tid % W2;
  int t1 = tid / W2;
  int y  = t1 % H2;
  int t2 = t1 / H2;
  int cg = t2 % 16;
  int bi = t2 / 16;
  const float* src = fm + (((size_t)bi * DCH + cg * 8) * H2 + y) * W2 + x;
  h8 v;
#pragma unroll
  for (int j = 0; j < 8; j++)
    v[j] = (_Float16)src[(size_t)j * H2 * W2];
  *(h8*)(ft + (((size_t)bi * H2 + y) * W2 + x) * DCH + cg * 8) = v;
}

// ---------------------------------------------------------------------------
// Pass 1b: feat1 f32 -> f16, plus zero-fill the guard pixel.
// ---------------------------------------------------------------------------
__global__ __launch_bounds__(256) void f1_convert_kernel(
    const float* __restrict__ f1, _Float16* __restrict__ ws) {
  int tid = blockIdx.x * 256 + threadIdx.x;       // 65536 threads, 8 elems each
  if (tid < (NB * NPTS * DCH) / 8) {
    const float* src = f1 + (size_t)tid * 8;
    h8 v;
#pragma unroll
    for (int j = 0; j < 8; j++) v[j] = (_Float16)src[j];
    *(h8*)(ws + F1H_OFF + (size_t)tid * 8) = v;
  }
  if (tid < DCH / 8) {                            // 16 threads zero guard pixel
    h8 z;
#pragma unroll
    for (int j = 0; j < 8; j++) z[j] = (_Float16)0.0f;
    *(h8*)(ws + (size_t)ZPIX * DCH + (size_t)tid * 8) = z;
  }
}

__device__ __forceinline__ void load_frags(h8 (&f)[8], const h8* p, int goff) {
#pragma unroll
  for (int q = 0; q < 4; q++) {
    f[2 * q]     = p[q * 4 + goff];
    f[2 * q + 1] = p[q * 4 + goff + 2];
  }
}

// ---------------------------------------------------------------------------
// Pass 2: one wave (32 lanes) per point.
//   Phase A: G[pix] = feat1[p] . FT[pixel]  via v_wmma_f32_16x16x32_f16.
//            Explicit 2-stage software pipeline over tile pairs: the next
//            pair's 16 b128 loads are issued (into cross-iteration live
//            registers) before the current pair's two interleaved WMMA
//            chains, forcing loads to stay in flight across the matrix ops.
//   Phase B: separable bilinear on scalar G, softmax, coordinate moments.
// ---------------------------------------------------------------------------
__global__ __launch_bounds__(256, 1) void caps_window_kernel(
    const float* __restrict__ coord2,
    const _Float16* __restrict__ ws,
    float* __restrict__ out) {
  __shared__ float Gs[8 * GSTRIDE];

  const int lane = threadIdx.x & 31;
  const int w    = threadIdx.x >> 5;
  const int wid  = blockIdx.x * 8 + w;            // 0..4095, exact
  const int bi   = wid >> 11;

  const float cx = coord2[(size_t)wid * 2 + 0];
  const float cy = coord2[(size_t)wid * 2 + 1];
  // patch origin = floor of first window sample's pixel coordinate
  const int x0 = (int)floorf((cx - 0.125f + 1.0f) * 80.0f - 0.5f);
  const int y0 = (int)floorf((cy - 0.125f + 1.0f) * 60.0f - 0.5f);

  // prefetch first patch row (global_prefetch_b8)
  {
    int yc = y0 < 0 ? 0 : (y0 >= H2 ? H2 - 1 : y0);
    int xc = x0 + lane; xc = xc < 0 ? 0 : (xc >= W2 ? W2 - 1 : xc);
    __builtin_prefetch(ws + ((size_t)((bi * H2 + yc) * W2 + xc)) * DCH, 0, 1);
  }

  // B fragments: feat1 channel chunk replicated across N.
  // B layout (32x16 f16): lanes 0-15 hold K=0..7,16..23; lanes 16-31 K=8..15,24..31.
  const h8* fp = (const h8*)(ws + F1H_OFF + (size_t)wid * DCH);
  const int gb8 = (lane < 16) ? 0 : 1;            // h8-chunk select within 32ch
  v16h bfrag[4];
#pragma unroll
  for (int q = 0; q < 4; q++) {
    h8 blo = fp[q * 4 + gb8];
    h8 bhi = fp[q * 4 + gb8 + 2];
    bfrag[q] = __builtin_shufflevector(blo, bhi,
        0, 1, 2, 3, 4, 5, 6, 7, 8, 9, 10, 11, 12, 13, 14, 15);
  }

  float* G = Gs + w * GSTRIDE;
  const int m    = lane & 15;
  const int goff = (lane < 16) ? 0 : 1;           // h8-chunk select within 32ch
  const int sbase = (bi * H2 + y0) * W2 + x0;     // wave-uniform

  // incremental patch coordinates: pix = t*16 + m ; r = pix/22, c = pix%22.
  // Stepping past the patch makes r >= 17 -> invalid -> guard pixel, so the
  // pipeline prologue overhang needs no special casing.
  int r = 0, c = m;
  auto next_ptr = [&]() -> const h8* {
    int xx = x0 + c, yy = y0 + r;
    int v = (int)((r < PATCH_H) & ((unsigned)xx < (unsigned)W2) &
                  ((unsigned)yy < (unsigned)H2));
    int lin = sbase + r * W2 + c;
    int pidx = ZPIX + (lin - ZPIX) * v;           // branchless zero-pad select
    const h8* p = (const h8*)(ws + (size_t)pidx * DCH);
    c += 16; int g = (c >= PATCH_W); c -= g * PATCH_W; r += g;
    return p;
  };

  // pipeline prologue: load pair 0
  h8 cA[8], cB[8];
  load_frags(cA, next_ptr(), goff);
  load_frags(cB, next_ptr(), goff);

#pragma unroll 2
  for (int tt = 0; tt < TILES / 2; tt++) {
    // issue next pair's loads first (kept live into the next iteration)
    h8 nA[8], nB[8];
    load_frags(nA, next_ptr(), goff);
    load_frags(nB, next_ptr(), goff);

    // two independent interleaved WMMA chains on the current pair
    v8f acc0 = {}, acc1 = {};
#pragma unroll
    for (int q = 0; q < 4; q++) {
      v16h aA = __builtin_shufflevector(cA[2 * q], cA[2 * q + 1],
          0, 1, 2, 3, 4, 5, 6, 7, 8, 9, 10, 11, 12, 13, 14, 15);
      v16h aB = __builtin_shufflevector(cB[2 * q], cB[2 * q + 1],
          0, 1, 2, 3, 4, 5, 6, 7, 8, 9, 10, 11, 12, 13, 14, 15);
      acc0 = __builtin_amdgcn_wmma_f32_16x16x32_f16(
          false, aA, false, bfrag[q], (short)0, acc0, false, false);
      acc1 = __builtin_amdgcn_wmma_f32_16x16x32_f16(
          false, aB, false, bfrag[q], (short)0, acc1, false, false);
    }

    // D col 0: lane0 holds M=0..7, lane16 holds M=8..15
    if ((lane & 15) == 0) {
      float* dst = G + (2 * tt) * 16 + ((lane >> 4) << 3);
#pragma unroll
      for (int e = 0; e < 8; e++) dst[e] = acc0[e];
#pragma unroll
      for (int e = 0; e < 8; e++) dst[16 + e] = acc1[e];
    }

    // rotate buffers (pure register renaming under unroll)
#pragma unroll
    for (int i = 0; i < 8; i++) { cA[i] = nA[i]; cB[i] = nB[i]; }
  }
  __syncthreads();

  // -------- Phase B: bilinear on G, softmax, moments (10 samples / lane) ----
  const float sx = 0.25f / 19.0f, sy = 0.25f / 14.0f;
  float sims[10];
  float mx = -3.0e38f;
#pragma unroll
  for (int it = 0; it < 10; it++) {
    int k = lane + it * 32;
    float s = -3.0e38f;
    if (k < KWIN) {
      int j = k / LEN_W;
      int i = k - j * LEN_W;
      float xsv = -0.125f + (float)i * sx;
      float ysv = -0.125f + (float)j * sy;
      float px = (cx + xsv + 1.0f) * 80.0f - 0.5f;
      float py = (cy + ysv + 1.0f) * 60.0f - 0.5f;
      float fx = floorf(px), fy = floorf(py);
      float wx = px - fx, wy = py - fy;
      int cc = (int)fx - x0; cc = cc < 0 ? 0 : (cc > PATCH_W - 2 ? PATCH_W - 2 : cc);
      int rr = (int)fy - y0; rr = rr < 0 ? 0 : (rr > PATCH_H - 2 ? PATCH_H - 2 : rr);
      const float* g0 = G + rr * PATCH_W + cc;
      float g00 = g0[0], g01 = g0[1], g10 = g0[PATCH_W], g11 = g0[PATCH_W + 1];
      s = g00 * (1.0f - wx) * (1.0f - wy) + g01 * wx * (1.0f - wy)
        + g10 * (1.0f - wx) * wy          + g11 * wx * wy;
    }
    sims[it] = s;
    mx = fmaxf(mx, s);
  }
#pragma unroll
  for (int o = 16; o > 0; o >>= 1) mx = fmaxf(mx, __shfl_xor(mx, o, 32));

  float S = 0.f, ex = 0.f, ey = 0.f, exx = 0.f, eyy = 0.f;
#pragma unroll
  for (int it = 0; it < 10; it++) {
    int k = lane + it * 32;
    if (k < KWIN) {
      int j = k / LEN_W;
      int i = k - j * LEN_W;
      float gxv = cx + (-0.125f + (float)i * sx);
      float gyv = cy + (-0.125f + (float)j * sy);
      float p = __expf(sims[it] - mx);
      S += p; ex += p * gxv; ey += p * gyv;
      exx += p * gxv * gxv; eyy += p * gyv * gyv;
    }
  }
#pragma unroll
  for (int o = 16; o > 0; o >>= 1) {
    S   += __shfl_xor(S,   o, 32);
    ex  += __shfl_xor(ex,  o, 32);
    ey  += __shfl_xor(ey,  o, 32);
    exx += __shfl_xor(exx, o, 32);
    eyy += __shfl_xor(eyy, o, 32);
  }
  if (lane == 0) {
    float inv = 1.0f / S;
    float mex = ex * inv, mey = ey * inv;
    float vx = exx * inv - mex * mex;
    float vy = eyy * inv - mey * mey;
    vx = vx < 1e-10f ? 1e-10f : vx;
    vy = vy < 1e-10f ? 1e-10f : vy;
    out[(size_t)wid * 2 + 0] = mex;
    out[(size_t)wid * 2 + 1] = mey;
    out[NB * NPTS * 2 + wid] = sqrtf(vx) + sqrtf(vy);
  }
}

extern "C" void kernel_launch(void* const* d_in, const int* in_sizes, int n_in,
                              void* d_out, int out_size, void* d_ws, size_t ws_size,
                              hipStream_t stream) {
  const float* feat1    = (const float*)d_in[0];
  const float* featmap2 = (const float*)d_in[1];
  const float* coord2   = (const float*)d_in[2];
  _Float16* ws = (_Float16*)d_ws;   // FT (9.83MB) + guard pixel + feat1-f16 (1MB)

  int tp_threads = NB * 16 * H2 * W2;             // 614400
  ft_transpose_kernel<<<(tp_threads + 255) / 256, 256, 0, stream>>>(featmap2, ws);

  int cv_threads = (NB * NPTS * DCH) / 8;         // 65536
  f1_convert_kernel<<<(cv_threads + 255) / 256, 256, 0, stream>>>(feat1, ws);

  // 4096 points, 1 wave each, 8 waves per 256-thread block -> 512 blocks exact
  caps_window_kernel<<<(NB * NPTS) / 8, 256, 0, stream>>>(
      coord2, ws, (float*)d_out);
}
